// WeightsAttention_18889266168042
// MI455X (gfx1250) — matrix operand order repaired and checked by reading, not compiled, not run on hardware
//
#include <hip/hip_runtime.h>

// ---------------------------------------------------------------------------
// CDNA5 (gfx1250) sparsemax memory-attention pipeline.
// Heavy math: v_wmma_f32_16x16x32_bf16 (bf16 in, f32 acc).
// Global->LDS staging: GLOBAL_LOAD_ASYNC_TO_LDS_B128 (ASYNCcnt) + LDS double
// buffering, so the DMA path overlaps with WMMA issue.
// ---------------------------------------------------------------------------

typedef __attribute__((ext_vector_type(16))) __bf16 v16bf;
typedef __attribute__((ext_vector_type(8)))  float  v8f;

union Frag16 { v16bf v; uint4 q[2]; };

#define EPI_F32        0   // store f32
#define EPI_BF16       1   // store bf16
#define EPI_BIAS_RELU  2   // +bias, relu, store bf16
#define EPI_BIAS_F32   3   // +bias, store f32

// Async copy of one 16B chunk from global memory directly into LDS.
// VDST = per-lane LDS byte offset, VADDR = per-lane 64-bit global address.
// Tracked with ASYNCcnt; completion is in-order among async loads.
__device__ __forceinline__ void async_copy16(unsigned lds_off, const void* gptr) {
    asm volatile("global_load_async_to_lds_b128 %0, %1, off"
                 :: "v"(lds_off), "v"((unsigned long long)(uintptr_t)gptr)
                 : "memory");
}

// ---------------------------------------------------------------------------
// NT GEMM: C[M,N] = A[M,K] @ B[N,K]^T  (bf16 operands, f32 accumulate, WMMA)
// Block = 256 threads (8 wave32), tile 128(M) x 256(N), K-step 32.
// Waves in a 2x4 grid; each wave owns a 64x64 sub-tile = 4x4 WMMA tiles.
// M, K multiples of 128 / 32 (true at all call sites); N may be ragged.
// ---------------------------------------------------------------------------
template <int EPI>
__global__ __launch_bounds__(256) void gemm_nt_wmma(
    const __bf16* __restrict__ A, const __bf16* __restrict__ B,
    void* __restrict__ Cv, const float* __restrict__ bias,
    int M, int N, int K, int ldc, int cofs)
{
    __shared__ alignas(16) __bf16 sA[2][128 * 32];   // 2 x 8 KB
    __shared__ alignas(16) __bf16 sB[2][256 * 32];   // 2 x 16 KB

    const int tid  = threadIdx.x;
    const int lane = tid & 31;
    const int wave = tid >> 5;
    const int wm   = wave >> 2;  // 0..1  -> 64 rows each
    const int wn   = wave & 3;   // 0..3  -> 64 cols each
    const int bm0  = blockIdx.y * 128;
    const int bn0  = blockIdx.x * 256;

    v8f acc[4][4];
#pragma unroll
    for (int i = 0; i < 4; ++i)
#pragma unroll
        for (int j = 0; j < 4; ++j) acc[i][j] = (v8f)0.0f;

    const int mr = lane & 15;
    const int kh = lane >> 4;

    // Issue the 6 async 16B copies (2 for A, 4 for B) of one k-tile.
    auto stage_async = [&](int buf, int k0) {
#pragma unroll
        for (int i = 0; i < 2; ++i) {
            const int idx  = tid * 2 + i;  // 0..511
            const int row  = idx >> 2;     // 0..127
            const int quad = idx & 3;
            async_copy16((unsigned)(uintptr_t)&sA[buf][idx * 8],
                         A + (size_t)(bm0 + row) * K + k0 + quad * 8);
        }
#pragma unroll
        for (int i = 0; i < 4; ++i) {
            const int idx  = tid * 4 + i;  // 0..1023
            const int row  = idx >> 2;     // 0..255
            const int quad = idx & 3;
            int br = bn0 + row; if (br >= N) br = N - 1;  // clamp ragged N
            async_copy16((unsigned)(uintptr_t)&sB[buf][idx * 8],
                         B + (size_t)br * K + k0 + quad * 8);
        }
    };

    const int nsteps = K >> 5;
    stage_async(0, 0);  // prologue

    for (int s = 0; s < nsteps; ++s) {
        const int  cur      = s & 1;
        const bool has_next = (s + 1) < nsteps;
        if (has_next) {
            stage_async(cur ^ 1, (s + 1) << 5);
            // 6 async loads just issued for the next tile; in-order completion
            // means asynccnt<=6 implies the *current* tile has landed in LDS.
            asm volatile("s_wait_asynccnt 0x6" ::: "memory");
        } else {
            asm volatile("s_wait_asynccnt 0x0" ::: "memory");
        }
        __syncthreads();  // current tile visible to all waves

        const __bf16* a0 = sA[cur];
        const __bf16* b0 = sB[cur];

        // B fragments: 32x16 bf16 in contiguous k order (k = 16h + s), same as A
        // lane<16: N=lane, K0..15 ; lane>=16: N=lane-16, K16..31
        Frag16 bq[4];
#pragma unroll
        for (int tn = 0; tn < 4; ++tn) {
            const int c = wn * 64 + tn * 16 + mr;
            bq[tn].q[0] = *(const uint4*)&b0[c * 32 + kh * 16];
            bq[tn].q[1] = *(const uint4*)&b0[c * 32 + kh * 16 + 8];
        }
        // A fragments: 16x32 bf16 in contiguous k order (k = 16h + s), same as B
        // lane<16 (M=lane):    v0..7 = K0..15
        // lane>=16 (M=lane-16): v0..7 = K16..31
#pragma unroll
        for (int tm = 0; tm < 4; ++tm) {
            Frag16 af;
            const int r = wm * 64 + tm * 16 + mr;
            af.q[0] = *(const uint4*)&a0[r * 32 + kh * 16];
            af.q[1] = *(const uint4*)&a0[r * 32 + kh * 16 + 8];
#pragma unroll
            for (int tn = 0; tn < 4; ++tn)
                acc[tm][tn] = __builtin_amdgcn_wmma_f32_16x16x32_bf16(
                    false, af.v, false, bq[tn].v,
                    (short)0, acc[tm][tn], false, false);
        }
        __syncthreads();  // all reads of `cur` done -> buffer reusable
    }

    // Epilogue. C/D layout: VGPR r, lane<16 -> (M=r, N=lane);
    //                               lane>=16 -> (M=8+r, N=lane-16).
    const int nlo = lane & 15;
    const int mhi = (lane >> 4) * 8;
#pragma unroll
    for (int tm = 0; tm < 4; ++tm) {
#pragma unroll
        for (int tn = 0; tn < 4; ++tn) {
            const int gn = bn0 + wn * 64 + tn * 16 + nlo;
            if (gn >= N) continue;
            const int gm0 = bm0 + wm * 64 + tm * 16 + mhi;
            const float bv =
                (EPI == EPI_BIAS_RELU || EPI == EPI_BIAS_F32) ? bias[gn] : 0.0f;
#pragma unroll
            for (int r = 0; r < 8; ++r) {
                const float  val = acc[tm][tn][r];
                const size_t off = (size_t)(gm0 + r) * ldc + cofs + gn;
                if      (EPI == EPI_F32)       ((float*)Cv)[off]  = val;
                else if (EPI == EPI_BF16)      ((__bf16*)Cv)[off] = (__bf16)val;
                else if (EPI == EPI_BIAS_RELU) ((__bf16*)Cv)[off] = (__bf16)fmaxf(val + bv, 0.0f);
                else                           ((float*)Cv)[off]  = val + bv;
            }
        }
    }
}

// ---------------------------------------------------------------------------
// Row normalize (rsqrt(sum x^2 + eps)) + bf16 convert.
// One 256-thread block per row (dim = 1024).
// ---------------------------------------------------------------------------
__global__ __launch_bounds__(256) void normalize_enc(
    const float* __restrict__ X, __bf16* __restrict__ Xn,
    __bf16* __restrict__ Fin, int dim, int ldf)
{
    __shared__ float red[8];
    const int row = blockIdx.x, tid = threadIdx.x;
    const int lane = tid & 31, wv = tid >> 5;
    const float* x = X + (size_t)row * dim;
    const int per = dim / 256;  // 4
    float v[8];
    float ss = 0.0f;
    for (int i = 0; i < per; ++i) { v[i] = x[tid + i * 256]; ss += v[i] * v[i]; }
    for (int o = 16; o > 0; o >>= 1) ss += __shfl_xor(ss, o, 32);
    if (lane == 0) red[wv] = ss;
    __syncthreads();
    float tot = red[0];
    for (int j = 1; j < 8; ++j) tot += red[j];
    const float r = rsqrtf(tot + 1e-6f);
    for (int i = 0; i < per; ++i) {
        const int c = tid + i * 256;
        Xn[(size_t)row * dim + c]  = (__bf16)(v[i] * r);
        Fin[(size_t)row * ldf + c] = (__bf16)v[i];
    }
}

// Memory set: normalized row -> Yn [m,dim]; raw row scattered into the
// transposed buffer MT[dim, m] so GEMM2 can run in NT form.
__global__ __launch_bounds__(256) void normalize_mem(
    const float* __restrict__ X, __bf16* __restrict__ Yn,
    __bf16* __restrict__ MT, int dim, int mrows)
{
    __shared__ float red[8];
    const int row = blockIdx.x, tid = threadIdx.x;
    const int lane = tid & 31, wv = tid >> 5;
    const float* x = X + (size_t)row * dim;
    const int per = dim / 256;
    float v[8];
    float ss = 0.0f;
    for (int i = 0; i < per; ++i) { v[i] = x[tid + i * 256]; ss += v[i] * v[i]; }
    for (int o = 16; o > 0; o >>= 1) ss += __shfl_xor(ss, o, 32);
    if (lane == 0) red[wv] = ss;
    __syncthreads();
    float tot = red[0];
    for (int j = 1; j < 8; ++j) tot += red[j];
    const float r = rsqrtf(tot + 1e-6f);
    for (int i = 0; i < per; ++i) {
        const int c = tid + i * 256;
        Yn[(size_t)row * dim + c]   = (__bf16)(v[i] * r);
        MT[(size_t)c * mrows + row] = (__bf16)v[i];  // transpose
    }
}

__global__ __launch_bounds__(256) void f32_to_bf16(
    const float* __restrict__ in, __bf16* __restrict__ out, size_t n)
{
    size_t i = blockIdx.x * (size_t)blockDim.x + threadIdx.x;
    const size_t stride = gridDim.x * (size_t)blockDim.x;
    for (; i < n; i += stride) out[i] = (__bf16)in[i];
}

// ---------------------------------------------------------------------------
// Sparsemax over rows of S [b, m] via bisection on tau:
//   find tau s.t. sum(max(s - tau, 0)) == 1;  w = max(s - tau, 0)
// (sparsemax is shift-invariant, so sparsemax(-d) == sparsemax(s).)
// One 256-thread block per row; m/256 values per lane held in registers.
// Reductions: wave32 shfl_xor + 8-entry cross-wave LDS combine.
// ---------------------------------------------------------------------------
__global__ __launch_bounds__(256) void sparsemax_rows(
    const float* __restrict__ S, __bf16* __restrict__ W, int m)
{
    __shared__ float red[8];
    const int row = blockIdx.x, tid = threadIdx.x;
    const int lane = tid & 31, wv = tid >> 5;
    const float* s = S + (size_t)row * m;
    const int per = m / 256;  // 32
    float v[32];
    float mx = -3.0e38f;
    for (int i = 0; i < per; ++i) { v[i] = s[tid + i * 256]; mx = fmaxf(mx, v[i]); }
    for (int o = 16; o > 0; o >>= 1) mx = fmaxf(mx, __shfl_xor(mx, o, 32));
    if (lane == 0) red[wv] = mx;
    __syncthreads();
    mx = red[0];
    for (int j = 1; j < 8; ++j) mx = fmaxf(mx, red[j]);
    __syncthreads();

    float lo = mx - 1.0f, hi = mx;  // sum(lo) >= 1, sum(hi) == 0
    for (int it = 0; it < 40; ++it) {
        const float tau = 0.5f * (lo + hi);
        float sum = 0.0f;
        for (int i = 0; i < per; ++i) sum += fmaxf(v[i] - tau, 0.0f);
        for (int o = 16; o > 0; o >>= 1) sum += __shfl_xor(sum, o, 32);
        if (lane == 0) red[wv] = sum;
        __syncthreads();
        float tot = red[0];
        for (int j = 1; j < 8; ++j) tot += red[j];
        __syncthreads();
        if (tot >= 1.0f) lo = tau; else hi = tau;
    }
    const float tau = 0.5f * (lo + hi);
    for (int i = 0; i < per; ++i)
        W[(size_t)row * m + tid + i * 256] = (__bf16)fmaxf(v[i] - tau, 0.0f);
}

// ---------------------------------------------------------------------------
extern "C" void kernel_launch(void* const* d_in, const int* in_sizes, int n_in,
                              void* d_out, int out_size, void* d_ws, size_t ws_size,
                              hipStream_t stream)
{
    const float* enc = (const float*)d_in[0];
    const float* mem = (const float*)d_in[1];
    // d_in[2] = loss_weights (unused by the reference computation)
    const float* W1 = (const float*)d_in[3];
    const float* b1 = (const float*)d_in[4];
    const float* W2 = (const float*)d_in[5];
    const float* b2 = (const float*)d_in[6];

    const int dim  = 1024;
    const int b    = in_sizes[0] / dim;  // 4096
    const int m    = in_sizes[1] / dim;  // 8192
    const int hdim = in_sizes[4];        // 4096
    const int odim = in_sizes[6];        // 1000
    const int fdim = 2 * dim;            // 2048

    char* p = (char*)d_ws;
    auto take = [&](size_t bytes) {
        char* r = p;
        p += (bytes + 255) & ~(size_t)255;
        return r;
    };
    __bf16* Xn  = (__bf16*)take((size_t)b * dim * 2);     // normalized encoder
    __bf16* Yn  = (__bf16*)take((size_t)m * dim * 2);     // normalized memory
    __bf16* MT  = (__bf16*)take((size_t)dim * m * 2);     // memory_set^T (raw)
    __bf16* Fin = (__bf16*)take((size_t)b * fdim * 2);    // [enc | mem_vec] bf16
    __bf16* W1b = (__bf16*)take((size_t)hdim * fdim * 2);
    __bf16* W2b = (__bf16*)take((size_t)odim * hdim * 2);
    __bf16* Hb  = (__bf16*)take((size_t)b * hdim * 2);    // relu hidden, bf16
    __bf16* Wt  = (__bf16*)take((size_t)b * m * 2);       // sparsemax weights
    float*  S   = (float*)take((size_t)b * m * 4);        // similarity scores

    // Precision conversions / normalization
    normalize_enc<<<b, 256, 0, stream>>>(enc, Xn, Fin, dim, fdim);
    normalize_mem<<<m, 256, 0, stream>>>(mem, Yn, MT, dim, m);
    f32_to_bf16<<<2048, 256, 0, stream>>>(W1, W1b, (size_t)hdim * fdim);
    f32_to_bf16<<<2048, 256, 0, stream>>>(W2, W2b, (size_t)odim * hdim);

    // GEMM1: S = Xn @ Yn^T   [b, m], f32 scores
    gemm_nt_wmma<EPI_F32><<<dim3(m / 256, b / 128), 256, 0, stream>>>(
        Xn, Yn, S, nullptr, b, m, dim, m, 0);

    // Sparsemax rows: Wt = sparsemax(S)
    sparsemax_rows<<<b, 256, 0, stream>>>(S, Wt, m);

    // GEMM2: Fin[:, dim:2dim] = Wt @ memory_set  (NT against MT [dim, m])
    gemm_nt_wmma<EPI_BF16><<<dim3(dim / 256, b / 128), 256, 0, stream>>>(
        Wt, MT, Fin, nullptr, b, dim, m, fdim, dim);

    // GEMM3: Hb = relu(Fin @ W1^T + b1)   [b, hdim]
    gemm_nt_wmma<EPI_BIAS_RELU><<<dim3(hdim / 256, b / 128), 256, 0, stream>>>(
        Fin, W1b, Hb, b1, b, hdim, fdim, hdim, 0);

    // GEMM4: out = Hb @ W2^T + b2   [b, odim], f32
    gemm_nt_wmma<EPI_BIAS_F32><<<dim3((odim + 255) / 256, b / 128), 256, 0, stream>>>(
        Hb, W2b, d_out, b2, b, odim, hdim, odim, 0);
}
